// EnsembleE2EModule_19756849562163
// MI455X (gfx1250) — compile-verified, matrix operand order checked by reading, not run on hardware
//
#include <hip/hip_runtime.h>
#include <hip/hip_bf16.h>
#include <math.h>

// ---------------- types ----------------
typedef __attribute__((ext_vector_type(16))) __bf16 v16bf;
typedef __attribute__((ext_vector_type(8)))  __bf16 bf16x8;
typedef __attribute__((ext_vector_type(8)))  float  v8f;

union Frag  { v16bf v; bf16x8 h[2]; };
union FAcc  { v8f v; float f[8]; };

#define B_ROWS 1024
#define DDIM   512
#define E_EXP  128
#define C_OUT  256
#define BPAD   520   // padded LDS row stride (bf16): 1040B -> 260 mod 64 = 4 banks/lane step

// ---------------- helpers ----------------
__device__ __forceinline__ float wave_sum(float v) {
  #pragma unroll
  for (int o = 16; o; o >>= 1) v += __shfl_xor(v, o, 32);
  return v;
}
__device__ __forceinline__ float wave_max(float v) {
  #pragma unroll
  for (int o = 16; o; o >>= 1) v = fmaxf(v, __shfl_xor(v, o, 32));
  return v;
}

// CDNA5 async global->LDS copy (16B per lane), tracked by ASYNCcnt.
// LDS offset = low 32 bits of the flat shared-memory address.
__device__ __forceinline__ void async_copy_b128(void* lds_dst, const void* gsrc) {
  asm volatile("global_load_async_to_lds_b128 %0, %1, off"
               :: "v"((unsigned)(uintptr_t)lds_dst),
                  "v"((unsigned long long)(uintptr_t)gsrc)
               : "memory");
}
__device__ __forceinline__ void wait_async0() {
  asm volatile("s_wait_asynccnt 0x0" ::: "memory");
}

// ---------------- fp32 -> bf16 (8 elems / thread) ----------------
__global__ void cvt_bf16_kernel(const float* __restrict__ s, __bf16* __restrict__ d, int n8) {
  int i = blockIdx.x * blockDim.x + threadIdx.x;
  if (i >= n8) return;
  const float4* s4 = (const float4*)s;
  float4 a = s4[2 * i], b = s4[2 * i + 1];
  bf16x8 o = { (__bf16)a.x, (__bf16)a.y, (__bf16)a.z, (__bf16)a.w,
               (__bf16)b.x, (__bf16)b.y, (__bf16)b.z, (__bf16)b.w };
  ((bf16x8*)d)[i] = o;
}

// ---------------- per-row 1/||x|| ----------------
__global__ __launch_bounds__(32) void rownorm_kernel(const float* __restrict__ x,
                                                     float* __restrict__ invn) {
  int b = blockIdx.x;
  const float* r = x + (size_t)b * DDIM;
  float s = 0.f;
  for (int i = threadIdx.x; i < DDIM; i += 32) { float v = r[i]; s += v * v; }
  s = wave_sum(s);
  if (threadIdx.x == 0) invn[b] = 1.0f / fmaxf(sqrtf(s), 1e-12f);
}

// ---------------- generic 16x16-tile WMMA GEMM over D=512 ----------------
// MODE 0: cos head   -> out0 = 1 - z*invn[r] (cos_dist), out1 = z*invn[r] (cos_sim)
// MODE 1: tanh head  -> out0 = tanh((z+bias[c])/10)*10
// MODE 2: raw logits -> out0 = z + bias[c]
template <int MODE, int N>
__global__ __launch_bounds__(32) void gemm_head_kernel(const __bf16* __restrict__ A,
                                                       const __bf16* __restrict__ W,
                                                       const float* __restrict__ invn,
                                                       const float* __restrict__ bias,
                                                       float* __restrict__ out0,
                                                       float* __restrict__ out1) {
  int tm = blockIdx.x * 16, tn = blockIdx.y * 16;
  int lane = threadIdx.x, hl = lane >> 4, lr = lane & 15;
  const __bf16* arow = A + (size_t)(tm + lr) * DDIM;
  const __bf16* brow = W + (size_t)(tn + lr) * DDIM;
  FAcc acc; acc.v = {};
  #pragma unroll
  for (int k = 0; k < DDIM / 32; ++k) {
    int kk = k * 32 + hl * 8;
    Frag a, b;
    a.h[0] = *(const bf16x8*)(arow + kk);
    a.h[1] = *(const bf16x8*)(arow + kk + 16);
    b.h[0] = *(const bf16x8*)(brow + kk);
    b.h[1] = *(const bf16x8*)(brow + kk + 16);
    acc.v = __builtin_amdgcn_wmma_f32_16x16x32_bf16(false, a.v, false, b.v,
                                                    (short)0, acc.v, false, false);
  }
  int c = tn + lr;
  #pragma unroll
  for (int i = 0; i < 8; ++i) {
    int r = tm + i + 8 * hl;
    float z = acc.f[i];
    if (MODE == 0) {
      float cs = z * invn[r];
      out0[(size_t)r * N + c] = 1.0f - cs;
      out1[(size_t)r * N + c] = cs;
    } else if (MODE == 1) {
      out0[(size_t)r * N + c] = tanhf((z + bias[c]) * 0.1f) * 10.0f;
    } else {
      out0[(size_t)r * N + c] = z + bias[c];
    }
  }
}

// ---------------- log_softmax over 256 cols, one wave per row ----------------
__global__ __launch_bounds__(32) void logsoftmax_kernel(const float* __restrict__ z,
                                                        float* __restrict__ out) {
  int b = blockIdx.x;
  const float* r = z + (size_t)b * C_OUT;
  float v[8]; float m = -INFINITY;
  #pragma unroll
  for (int i = 0; i < 8; ++i) { v[i] = r[threadIdx.x + 32 * i]; m = fmaxf(m, v[i]); }
  m = wave_max(m);
  float s = 0.f;
  #pragma unroll
  for (int i = 0; i < 8; ++i) s += expf(v[i] - m);
  s = wave_sum(s);
  float lse = m + logf(s);
  #pragma unroll
  for (int i = 0; i < 8; ++i) out[(size_t)b * C_OUT + threadIdx.x + 32 * i] = v[i] - lse;
}

// ---------------- Sinkhorn soft-top-k: one wave per row, N=128 (4/lane) ------
__global__ __launch_bounds__(32) void sinkhorn_kernel(const float* __restrict__ cdist,
                                                      const float* __restrict__ csim,
                                                      float* __restrict__ knn_out,
                                                      float* __restrict__ wnum,
                                                      float* __restrict__ den) {
  const float eps = 5e-4f, inv_eps = 2000.0f;
  const float log_mu = -logf(128.0f);
  const float lnu0 = logf(16.0f / 128.0f), lnu1 = logf(112.0f / 128.0f);
  int b = blockIdx.x, lane = threadIdx.x;
  const float* dd = cdist + (size_t)b * E_EXP;

  float d0[4], C0[4], C1[4], f[4];
  float mx = -INFINITY;
  #pragma unroll
  for (int j = 0; j < 4; ++j) { d0[j] = dd[lane + 32 * j]; mx = fmaxf(mx, d0[j]); }
  mx = fmaxf(wave_max(mx), 1e-12f);
  #pragma unroll
  for (int j = 0; j < 4; ++j) {
    float dn = d0[j] / mx;
    C0[j] = dn * dn;
    float t = dn - 1.0f;
    C1[j] = t * t;
    f[j] = 0.0f;
  }
  float g0 = 0.f, g1 = 0.f;

  for (int it = 0; it < 400; ++it) {
    float t0[4], t1[4], m0 = -INFINITY, m1 = -INFINITY;
    #pragma unroll
    for (int j = 0; j < 4; ++j) {
      t0[j] = (f[j] - C0[j]) * inv_eps; m0 = fmaxf(m0, t0[j]);
      t1[j] = (f[j] - C1[j]) * inv_eps; m1 = fmaxf(m1, t1[j]);
    }
    m0 = wave_max(m0); m1 = wave_max(m1);
    float s0 = 0.f, s1 = 0.f;
    #pragma unroll
    for (int j = 0; j < 4; ++j) { s0 += expf(t0[j] - m0); s1 += expf(t1[j] - m1); }
    s0 = wave_sum(s0); s1 = wave_sum(s1);
    g0 = eps * lnu0 - eps * (m0 + logf(s0));
    g1 = eps * lnu1 - eps * (m1 + logf(s1));
    #pragma unroll
    for (int j = 0; j < 4; ++j) {
      float u0 = (g0 - C0[j]) * inv_eps, u1 = (g1 - C1[j]) * inv_eps;
      float m = fmaxf(u0, u1);
      f[j] = eps * log_mu - eps * (m + logf(expf(u0 - m) + expf(u1 - m)));
    }
  }

  float dsum = 0.f;
  #pragma unroll
  for (int j = 0; j < 4; ++j) {
    int idx = lane + 32 * j;
    float k = expf((f[j] + g0 - C0[j]) * inv_eps) * 128.0f;   // Gamma[:,0]*N
    float cs = csim[(size_t)b * E_EXP + idx];
    knn_out[(size_t)b * E_EXP + idx] = k;
    wnum[(size_t)b * E_EXP + idx] = cs * k * k;                // gate^2 * cos
    dsum += cs * k;                                            // gate * cos
  }
  dsum = wave_sum(dsum);
  if (lane == 0) den[b] = dsum;
}

// ---------------- fused expert GEMM + tanh + ensemble reduction --------------
// Workgroup: 64 batch rows x 16 out cols, 4 waves. Each wave keeps its 16x512
// A tile in 128 VGPRs. All 4 waves share one 16x512 B tile per expert, moved
// global->LDS by GLOBAL_LOAD_ASYNC_TO_LDS_B128 (no data VGPRs -> no spills),
// double buffered with one s_wait_asynccnt + barrier per expert.
__global__ __launch_bounds__(128) void expert_kernel(const __bf16* __restrict__ xb,
                                                     const __bf16* __restrict__ ewb,
                                                     const float* __restrict__ eb,
                                                     const float* __restrict__ wnum,
                                                     const float* __restrict__ den,
                                                     float* __restrict__ out) {
  extern __shared__ char smem[];
  __bf16* Bs = (__bf16*)smem;                              // 2 * 16 * BPAD bf16 = 33,280 B
  float*  Ws = (float*)(smem + 2 * 16 * BPAD * 2);         // 64 * 128 f32     = 32,768 B
  int bm = blockIdx.x;          // 0..15 (64-row tile)
  int bn = blockIdx.y;          // 0..15 (16-col tile)
  int tid = threadIdx.x;
  int wave = tid >> 5, lane = tid & 31;
  int hl = lane >> 4, lr = lane & 15;

  // stage wnum tile (64 rows x 128 experts) into LDS
  {
    const float4* wg = (const float4*)(wnum + (size_t)bm * 64 * E_EXP);
    float4* ws4 = (float4*)Ws;
    for (int i = tid; i < 64 * E_EXP / 4; i += 128) ws4[i] = wg[i];
  }

  // per-wave register-resident A fragments (16 x 512 bf16 -> 16 frags)
  const __bf16* arow = xb + (size_t)(bm * 64 + wave * 16 + lr) * DDIM;
  Frag afr[16];
  #pragma unroll
  for (int k = 0; k < 16; ++k) {
    int kk = k * 32 + hl * 8;
    afr[k].h[0] = *(const bf16x8*)(arow + kk);
    afr[k].h[1] = *(const bf16x8*)(arow + kk + 16);
  }

  // B tile: 16 cols x 512 d = 1024 x 16B chunks; 128 threads x 8 async copies
  const uint4* bgbase = (const uint4*)(ewb + (size_t)bn * 16 * DDIM);
  int colc = tid & 63;          // 16B chunk within a row (64 per row)
  int rowc = tid >> 6;          // rows rowc + 2*i

  auto stage_b = [&](int e, int buf) {   // async global -> LDS, no VGPR data
    const uint4* bg = bgbase + (size_t)e * (C_OUT * DDIM / 8);
    __bf16* bs = Bs + buf * 16 * BPAD;
    #pragma unroll
    for (int i = 0; i < 8; ++i)
      async_copy_b128(bs + (rowc + 2 * i) * BPAD + colc * 8,
                      bg + (rowc + 2 * i) * 64 + colc);
  };

  stage_b(0, 0);
  wait_async0();
  __syncthreads();

  FAcc num; num.v = {};
  const float* ebcol = eb + bn * 16 + lr;
  for (int e = 0; e < E_EXP; ++e) {
    int cur = e & 1;
    if (e + 1 < E_EXP) stage_b(e + 1, cur ^ 1);   // async, overlaps WMMAs below

    const __bf16* brow = Bs + cur * 16 * BPAD + (size_t)lr * BPAD + hl * 8;
    Frag bfr[2];
    bfr[0].h[0] = *(const bf16x8*)(brow + 0);
    bfr[0].h[1] = *(const bf16x8*)(brow + 16);
    FAcc acc; acc.v = {};
    #pragma unroll
    for (int k = 0; k < 16; ++k) {
      if (k + 1 < 16) {                           // keep next frag pair in flight
        int kk = (k + 1) * 32;
        bfr[(k + 1) & 1].h[0] = *(const bf16x8*)(brow + kk);
        bfr[(k + 1) & 1].h[1] = *(const bf16x8*)(brow + kk + 16);
      }
      acc.v = __builtin_amdgcn_wmma_f32_16x16x32_bf16(false, afr[k].v, false, bfr[k & 1].v,
                                                      (short)0, acc.v, false, false);
    }
    float bias = ebcol[e * C_OUT];
    int rbase = wave * 16 + 8 * hl;
    #pragma unroll
    for (int i = 0; i < 8; ++i) {
      float w = Ws[(rbase + i) * E_EXP + e];
      num.f[i] += w * (tanhf((acc.f[i] + bias) * 0.1f) * 10.0f);
    }
    wait_async0();     // my async writes into Bs[cur^1] have landed
    __syncthreads();   // all waves done reading Bs[cur] / writing Bs[cur^1]
  }

  int colg = bn * 16 + lr;
  #pragma unroll
  for (int i = 0; i < 8; ++i) {
    int rg = bm * 64 + wave * 16 + 8 * hl + i;
    out[(size_t)rg * C_OUT + colg] = num.f[i] / den[rg];
  }
}

// ---------------- launch ----------------
extern "C" void kernel_launch(void* const* d_in, const int* in_sizes, int n_in,
                              void* d_out, int out_size, void* d_ws, size_t ws_size,
                              hipStream_t stream) {
  const float* x    = (const float*)d_in[0];
  const float* keys = (const float*)d_in[1];
  const float* ew   = (const float*)d_in[2];
  const float* eb   = (const float*)d_in[3];
  const float* vw   = (const float*)d_in[4];
  const float* vb   = (const float*)d_in[5];
  const float* tw   = (const float*)d_in[6];
  const float* tb   = (const float*)d_in[7];

  // output layout (floats): ensemble | tanh | vanilla | cos_dist | knn_sim
  float* out   = (float*)d_out;
  float* ens   = out;
  float* tanho = out + 262144;
  float* vano  = out + 524288;
  float* cdist = out + 786432;
  float* knno  = out + 917504;

  // workspace layout
  char* w = (char*)d_ws;
  __bf16* ewb  = (__bf16*)(w);                 // 128*256*512 bf16 = 32 MB
  __bf16* xbf  = (__bf16*)(w + 33554432);      // 1024*512 bf16
  __bf16* kbf  = (__bf16*)(w + 34603008);      // 128*512 bf16
  __bf16* vwb  = (__bf16*)(w + 34734080);      // 256*512 bf16
  __bf16* twb  = (__bf16*)(w + 34996224);      // 256*512 bf16
  float*  invn = (float*)(w + 35258368);       // 1024
  float*  csim = (float*)(w + 35262464);       // 1024*128
  float*  wnum = (float*)(w + 35786752);       // 1024*128
  float*  den  = (float*)(w + 36311040);       // 1024
  float*  vlog = (float*)(w + 36315136);       // 1024*256

  // 1) bf16 conversions
  cvt_bf16_kernel<<<(2097152 + 255) / 256, 256, 0, stream>>>(ew, ewb, 2097152);
  cvt_bf16_kernel<<<(65536   + 255) / 256, 256, 0, stream>>>(x,  xbf, 65536);
  cvt_bf16_kernel<<<(8192    + 255) / 256, 256, 0, stream>>>(keys, kbf, 8192);
  cvt_bf16_kernel<<<(16384   + 255) / 256, 256, 0, stream>>>(vw, vwb, 16384);
  cvt_bf16_kernel<<<(16384   + 255) / 256, 256, 0, stream>>>(tw, twb, 16384);

  // 2) row inverse norms
  rownorm_kernel<<<B_ROWS, 32, 0, stream>>>(x, invn);

  // 3) cos head (WMMA) -> cos_dist (out) + cos_sim (ws)
  gemm_head_kernel<0, 128><<<dim3(64, 8), 32, 0, stream>>>(xbf, kbf, invn, nullptr, cdist, csim);

  // 4) Sinkhorn soft-top-k -> knn (out), wnum/den (ws)
  sinkhorn_kernel<<<B_ROWS, 32, 0, stream>>>(cdist, csim, knno, wnum, den);

  // 5) tanh head (WMMA, fused epilogue) -> out
  gemm_head_kernel<1, 256><<<dim3(64, 16), 32, 0, stream>>>(xbf, twb, nullptr, tb, tanho, nullptr);

  // 6) vanilla head (WMMA) -> ws logits, then log_softmax -> out
  gemm_head_kernel<2, 256><<<dim3(64, 16), 32, 0, stream>>>(xbf, vwb, nullptr, vb, vlog, nullptr);
  logsoftmax_kernel<<<B_ROWS, 32, 0, stream>>>(vlog, vano);

  // 7) fused expert GEMM + tanh + weighted ensemble -> out (64x16 tiles, async-LDS B)
  expert_kernel<<<dim3(16, 16), 128, 66048, stream>>>(xbf, ewb, eb, wnum, den, ens);
}
